// Transformer_20126216749100
// MI455X (gfx1250) — matrix operand order repaired and checked by reading, not compiled, not run on hardware
//
#include <hip/hip_runtime.h>
#include <hip/hip_bf16.h>

// ---------------------------------------------------------------------------
// CDNA5 / gfx1250 transformer block forward.
//   h = emb[x] + pos_emb[s]
//   q,k   = h @ Wq/Wk               (bf16 WMMA GEMM, B pre-transposed)
//   vT    = (h @ Wv)^T              (same GEMM, transposed-store epilogue)
//   scores= q k^T / 32              (WMMA GEMM; B^T storage == k row-major!)
//   attn  = causal softmax          (bf16, zeros above diagonal)
//   z     = attn @ v + h            (WMMA GEMM, vT as B^T, f32 residual)
//   out   = z @ Wo + bo             (WMMA GEMM, dominant 4096x32000x1024)
// Heavy math through v_wmma_f32_16x16x32_bf16; LDS staging through
// global_load_async_to_lds_b128 + s_wait_asynccnt when available.
// ---------------------------------------------------------------------------

typedef __attribute__((ext_vector_type(16))) __bf16 v16bf;
typedef __attribute__((ext_vector_type(8)))  float  v8f;
typedef __attribute__((ext_vector_type(4)))  int    v4i;

#define T_B 2
#define T_S 2048
#define T_H 1024
#define T_V 32000
#define T_BS (T_B * T_S)

#if defined(__AMDGCN__) && __has_builtin(__builtin_amdgcn_global_load_async_to_lds_b128)
#define USE_ASYNC_LDS 1
#else
#define USE_ASYNC_LDS 0
#endif

// round-to-nearest-even f32 -> bf16 (raw ushort)
__device__ __forceinline__ unsigned short f2bfu(float f) {
  unsigned u = __float_as_uint(f);
  u += 0x7FFFu + ((u >> 16) & 1u);
  return (unsigned short)(u >> 16);
}
__device__ __forceinline__ __bf16 u2bf(unsigned short s) {
  __bf16 r;
  __builtin_memcpy(&r, &s, 2);
  return r;
}

struct __align__(16) U8 { unsigned short s[8]; };

__device__ __forceinline__ void wait_async0() {
#if USE_ASYNC_LDS
#if __has_builtin(__builtin_amdgcn_s_wait_asynccnt)
  __builtin_amdgcn_s_wait_asynccnt(0);
#else
  asm volatile("s_wait_asynccnt 0x0" ::: "memory");
#endif
#endif
}

#if USE_ASYNC_LDS
// one lane moves 16 bytes global -> LDS through the async path (ASYNCcnt)
__device__ __forceinline__ void async_cp16(unsigned short* l, const unsigned short* g) {
  __builtin_amdgcn_global_load_async_to_lds_b128(
      (__attribute__((address_space(1))) v4i*)const_cast<unsigned short*>(g),
      (__attribute__((address_space(3))) v4i*)l, 0, 0);
}
#endif

// ---------------------------------------------------------------------------
// Embedding + positional add. One block per (b,s) row, 256 threads, H=1024.
// ---------------------------------------------------------------------------
__global__ __launch_bounds__(256) void embed_kernel(
    const int* __restrict__ x, const float* __restrict__ emb,
    const float* __restrict__ pos, float* __restrict__ hf,
    unsigned short* __restrict__ hb) {
  const int row = blockIdx.x;             // b*S + s
  const int s   = row & (T_S - 1);
  const int tok = x[row];
  const int t   = threadIdx.x;            // 256 threads * 4 floats = 1024
  const float4* e = (const float4*)(emb + (size_t)tok * T_H);
  const float4* p = (const float4*)(pos + (size_t)s * T_H);
  float4 ev = e[t], pv = p[t];
  float4 r = make_float4(ev.x + pv.x, ev.y + pv.y, ev.z + pv.z, ev.w + pv.w);
  ((float4*)(hf + (size_t)row * T_H))[t] = r;
  unsigned short* ob = hb + (size_t)row * T_H + t * 4;
  ob[0] = f2bfu(r.x); ob[1] = f2bfu(r.y); ob[2] = f2bfu(r.z); ob[3] = f2bfu(r.w);
}

// ---------------------------------------------------------------------------
// Fused f32->bf16 convert + transpose: in[R][C] f32 -> out[C][R] bf16.
// 32x32 LDS tile, block (32,8).
// ---------------------------------------------------------------------------
__global__ __launch_bounds__(256) void convT_kernel(
    const float* __restrict__ in, unsigned short* __restrict__ out,
    int R, int C) {
  __shared__ float tile[32][33];
  const int bx = blockIdx.x * 32;   // col of in
  const int by = blockIdx.y * 32;   // row of in
  const int tx = threadIdx.x, ty = threadIdx.y;
#pragma unroll
  for (int i = 0; i < 32; i += 8)
    tile[ty + i][tx] = in[(size_t)(by + ty + i) * C + bx + tx];
  __syncthreads();
#pragma unroll
  for (int i = 0; i < 32; i += 8)
    out[(size_t)(bx + ty + i) * R + by + tx] = f2bfu(tile[tx][ty + i]);
}

// ---------------------------------------------------------------------------
// Causal softmax over scores row; writes bf16 attn (zeros beyond diagonal).
// ---------------------------------------------------------------------------
__global__ __launch_bounds__(256) void causal_softmax_kernel(
    const float* __restrict__ scores, unsigned short* __restrict__ attn) {
  const int row = blockIdx.x;            // 0 .. B*S-1
  const int s   = row & (T_S - 1);
  const int L   = s + 1;
  const float* sc = scores + (size_t)row * T_S;
  unsigned short* at = attn + (size_t)row * T_S;
  __shared__ float red[256];
  const int tid = threadIdx.x;

  float m = -3.4e38f;
  for (int j = tid; j < L; j += 256) m = fmaxf(m, sc[j]);
  red[tid] = m; __syncthreads();
  for (int off = 128; off > 0; off >>= 1) {
    if (tid < off) red[tid] = fmaxf(red[tid], red[tid + off]);
    __syncthreads();
  }
  m = red[0]; __syncthreads();

  float sum = 0.0f;
  for (int j = tid; j < L; j += 256) sum += __expf(sc[j] - m);
  red[tid] = sum; __syncthreads();
  for (int off = 128; off > 0; off >>= 1) {
    if (tid < off) red[tid] += red[tid + off];
    __syncthreads();
  }
  const float inv = 1.0f / red[0];

  for (int j = tid; j < T_S; j += 256) {
    float a = (j < L) ? __expf(sc[j] - m) * inv : 0.0f;
    at[j] = f2bfu(a);
  }
}

// ---------------------------------------------------------------------------
// bf16 WMMA GEMM with B in TRANSPOSED storage Bt[N][K] (row stride ldb):
//   C[M,N] = scale * (A[M,K] @ B[K,N]) (+bias) (+resid)
// Block tile 128x128, BK=32, 256 thr = 8 waves, wave = 64x32 (4x2 WMMA tiles).
// Double-buffered LDS; staging via async-to-LDS b128 when available.
// ---------------------------------------------------------------------------
#define BM 128
#define BN 128
#define BK 32
#define LPAD 40   // LDS row stride in halves (80B, 16B-aligned, pads banks)

template <int OUT_F32, int OUT_BF16, int OUT_T, int RESID, int BIAS>
__global__ __launch_bounds__(256) void gemm_bf16_wmma(
    const unsigned short* __restrict__ A, const unsigned short* __restrict__ Bt,
    float* __restrict__ Cf, unsigned short* __restrict__ Cb,
    const float* __restrict__ resid, const float* __restrict__ bias,
    int M, int N, int K, int ldb, float scale) {
  __shared__ __align__(16) unsigned short As[2][BM * LPAD];  // [row][k]
  __shared__ __align__(16) unsigned short Bs[2][BN * LPAD];  // [col][k]

  const int tid  = threadIdx.x;
  const int lane = tid & 31;
  const int wave = tid >> 5;
  const int wm   = wave & 1;       // 64-row slab
  const int wn   = wave >> 1;      // 32-col slab
  const int lr   = lane & 15;
  const int lg   = lane >> 4;

  const int row0 = blockIdx.y * BM;
  const int col0 = blockIdx.x * BN;

  v8f acc[4][2] = {};

  // stage tile kb into LDS buffer `buf`: A[BMxBK] and Bt[BNxBK], both K-major
  auto stage = [&](int kb, int buf) {
    const int k0 = kb * BK;
    unsigned short* AsB = &As[buf][0];
    unsigned short* BsB = &Bs[buf][0];
#pragma unroll
    for (int it = 0; it < 2; ++it) {
      const int chunk = tid + it * 256;        // 0..511
      const int r  = chunk >> 2;               // row/col within tile
      const int c8 = (chunk & 3) * 8;          // 8-half group within BK
      const unsigned short* ga = A  + (size_t)(row0 + r) * K   + k0 + c8;
      const unsigned short* gb = Bt + (size_t)(col0 + r) * ldb + k0 + c8;
#if USE_ASYNC_LDS
      async_cp16(&AsB[r * LPAD + c8], ga);
      async_cp16(&BsB[r * LPAD + c8], gb);
#else
      const U8 va = *reinterpret_cast<const U8*>(ga);
      const U8 vb = *reinterpret_cast<const U8*>(gb);
      *reinterpret_cast<U8*>(&AsB[r * LPAD + c8]) = va;
      *reinterpret_cast<U8*>(&BsB[r * LPAD + c8]) = vb;
#endif
    }
  };

  const int nk = K / BK;
  stage(0, 0);
  for (int kb = 0; kb < nk; ++kb) {
    wait_async0();
    __syncthreads();                 // tile kb visible; prev compute done
    if (kb + 1 < nk) stage(kb + 1, (kb + 1) & 1);

    const unsigned short* AsB = &As[kb & 1][0];
    const unsigned short* BsB = &Bs[kb & 1][0];

    // ---- B fragments: 32x16 bf16; lane = N%16, halves carry K per lane grp
    v16bf bfr[2];
#pragma unroll
    for (int sn = 0; sn < 2; ++sn) {
      const int ccol = wn * 32 + sn * 16 + lr;
#pragma unroll
      for (int j = 0; j < 16; ++j)
        bfr[sn][j] = u2bf(BsB[ccol * LPAD + lg * 8 + j + (j & 8)]);
    }
    // ---- A fragments + 8 WMMAs ----
#pragma unroll
    for (int sm = 0; sm < 4; ++sm) {
      const int rrow = wm * 64 + sm * 16 + lr;
      v16bf af;
#pragma unroll
      for (int j = 0; j < 8; ++j) {
        af[j]     = u2bf(AsB[rrow * LPAD + lg * 8 + j]);
        af[j + 8] = u2bf(AsB[rrow * LPAD + lg * 8 + 16 + j]);
      }
#pragma unroll
      for (int sn = 0; sn < 2; ++sn)
        acc[sm][sn] = __builtin_amdgcn_wmma_f32_16x16x32_bf16(
            false, af, false, bfr[sn], (short)0, acc[sm][sn], false, false);
    }
  }

  // ---- epilogue: C layout  M = i + 8*lg, N = lr  per 16x16 tile ----
#pragma unroll
  for (int sm = 0; sm < 4; ++sm) {
#pragma unroll
    for (int sn = 0; sn < 2; ++sn) {
      const int gcol = col0 + wn * 32 + sn * 16 + lr;
#pragma unroll
      for (int i = 0; i < 8; ++i) {
        const int grow = row0 + wm * 64 + sm * 16 + i + lg * 8;
        float v = acc[sm][sn][i] * scale;
        if (BIAS)  v += bias[gcol];
        if (RESID) v += resid[(size_t)grow * N + gcol];
        if (OUT_F32)  Cf[(size_t)grow * N + gcol] = v;
        if (OUT_BF16) Cb[(size_t)grow * N + gcol] = f2bfu(v);
        if (OUT_T)    Cb[(size_t)gcol * M + grow] = f2bfu(v);  // C^T store
      }
    }
  }
}

// ---------------------------------------------------------------------------
// Host-side launcher
// ---------------------------------------------------------------------------
extern "C" void kernel_launch(void* const* d_in, const int* in_sizes, int n_in,
                              void* d_out, int out_size, void* d_ws, size_t ws_size,
                              hipStream_t stream) {
  (void)in_sizes; (void)n_in; (void)out_size; (void)ws_size;

  // setup_inputs order: x, emb, pos_emb, Wk, Wq, Wv, Wo, bo
  const int*   x   = (const int*)d_in[0];
  const float* emb = (const float*)d_in[1];
  const float* pos = (const float*)d_in[2];
  const float* Wk  = (const float*)d_in[3];
  const float* Wq  = (const float*)d_in[4];
  const float* Wv  = (const float*)d_in[5];
  const float* Wo  = (const float*)d_in[6];
  const float* bo  = (const float*)d_in[7];
  float* out = (float*)d_out;

  // ---- workspace carve-up (all chunks 256B aligned) ----
  char* w = (char*)d_ws;
  float*          h_f  = (float*)w;          w += (size_t)T_BS * T_H * 4;
  unsigned short* h_b  = (unsigned short*)w; w += (size_t)T_BS * T_H * 2;
  unsigned short* WqT  = (unsigned short*)w; w += (size_t)T_H * T_H * 2;
  unsigned short* WkT  = (unsigned short*)w; w += (size_t)T_H * T_H * 2;
  unsigned short* WvT  = (unsigned short*)w; w += (size_t)T_H * T_H * 2;
  unsigned short* WoT  = (unsigned short*)w; w += (size_t)T_H * T_V * 2;
  unsigned short* q_b  = (unsigned short*)w; w += (size_t)T_BS * T_H * 2;
  unsigned short* k_b  = (unsigned short*)w; w += (size_t)T_BS * T_H * 2;
  unsigned short* vT   = (unsigned short*)w; w += (size_t)T_BS * T_H * 2;  // [H][B*S]
  float*          sc   = (float*)w;          w += (size_t)T_B * T_S * T_S * 4;
  unsigned short* at   = (unsigned short*)w; w += (size_t)T_B * T_S * T_S * 2;
  unsigned short* z_b  = (unsigned short*)w; w += (size_t)T_BS * T_H * 2;

  // 1) embedding + positional (f32 residual + bf16 activation)
  embed_kernel<<<T_BS, 256, 0, stream>>>(x, emb, pos, h_f, h_b);

  // 2) weights: f32 [K][N] -> bf16 transposed [N][K]
  convT_kernel<<<dim3(T_H / 32, T_H / 32), dim3(32, 8), 0, stream>>>(Wq, WqT, T_H, T_H);
  convT_kernel<<<dim3(T_H / 32, T_H / 32), dim3(32, 8), 0, stream>>>(Wk, WkT, T_H, T_H);
  convT_kernel<<<dim3(T_H / 32, T_H / 32), dim3(32, 8), 0, stream>>>(Wv, WvT, T_H, T_H);
  convT_kernel<<<dim3(T_V / 32, T_H / 32), dim3(32, 8), 0, stream>>>(Wo, WoT, T_H, T_V);

  // 3) QKV projections: [4096,1024] @ [1024,1024]
  dim3 gqkv(T_H / BN, T_BS / BM);
  gemm_bf16_wmma<0, 1, 0, 0, 0><<<gqkv, 256, 0, stream>>>(
      h_b, WqT, (float*)nullptr, q_b, (const float*)nullptr,
      (const float*)nullptr, T_BS, T_H, T_H, T_H, 1.0f);
  gemm_bf16_wmma<0, 1, 0, 0, 0><<<gqkv, 256, 0, stream>>>(
      h_b, WkT, (float*)nullptr, k_b, (const float*)nullptr,
      (const float*)nullptr, T_BS, T_H, T_H, T_H, 1.0f);
  // v projection stored transposed: vT[H][B*S]
  gemm_bf16_wmma<0, 0, 1, 0, 0><<<gqkv, 256, 0, stream>>>(
      h_b, WvT, (float*)nullptr, vT, (const float*)nullptr,
      (const float*)nullptr, T_BS, T_H, T_H, T_H, 1.0f);

  for (int b = 0; b < T_B; ++b) {
    // 4) scores = q k^T / sqrt(H).  B^T storage of k^T is k itself: [S][H].
    gemm_bf16_wmma<1, 0, 0, 0, 0><<<dim3(T_S / BN, T_S / BM), 256, 0, stream>>>(
        q_b + (size_t)b * T_S * T_H, k_b + (size_t)b * T_S * T_H,
        sc + (size_t)b * T_S * T_S, (unsigned short*)nullptr,
        (const float*)nullptr, (const float*)nullptr,
        T_S, T_S, T_H, T_H, 0.03125f);
  }

  // 5) causal softmax -> bf16 attn (zeros above diagonal)
  causal_softmax_kernel<<<T_B * T_S, 256, 0, stream>>>(sc, at);

  for (int b = 0; b < T_B; ++b) {
    // 6) z = attn @ v + h.  B^T = vT (row stride B*S, batch column offset).
    gemm_bf16_wmma<0, 1, 0, 1, 0><<<dim3(T_H / BN, T_S / BM), 256, 0, stream>>>(
        at + (size_t)b * T_S * T_S, vT + (size_t)b * T_S,
        (float*)nullptr, z_b + (size_t)b * T_S * T_H,
        h_f + (size_t)b * T_S * T_H, (const float*)nullptr,
        T_S, T_H, T_S, T_BS, 1.0f);
  }

  // 7) out = z @ Wo + bo: [4096,1024] @ [1024,32000] -> f32 d_out
  gemm_bf16_wmma<1, 0, 0, 0, 1><<<dim3(T_V / BN, T_BS / BM), 256, 0, stream>>>(
      z_b, WoT, out, (unsigned short*)nullptr, (const float*)nullptr, bo,
      T_BS, T_V, T_H, T_H, 1.0f);
}
